// AttnProcessor_2800318677573
// MI455X (gfx1250) — matrix operand order repaired and checked by reading, not compiled, not run on hardware
//
#include <hip/hip_runtime.h>

// CDNA5 / gfx1250 wave32 WMMA attention.
// Pipeline: vectorized f32->f16 convert, f16 weight transpose (for contiguous
// WMMA B-fragment loads), Q/K/V projection GEMMs (4 N-tiles per wave, WMMA f16
// with f32 acc), flash-style online-softmax attention with 64-key blocks
// (16 WMMAs per softmax update), output GEMM + bias.

typedef __attribute__((ext_vector_type(16))) _Float16 v16h;
typedef __attribute__((ext_vector_type(8)))  _Float16 v8h;
typedef __attribute__((ext_vector_type(8)))  float    v8f;
typedef __attribute__((ext_vector_type(4)))  float    v4f;

#define S_LEN  4096
#define DMODEL 512
#define NHEADS 8
#define HDIM   64

static __device__ __forceinline__ v16h pack16(v8h lo, v8h hi) {
  return __builtin_shufflevector(lo, hi, 0,1,2,3,4,5,6,7,8,9,10,11,12,13,14,15);
}

// 8 elements per thread: 2x b128 in, 1x b128 out.
__global__ void f32_to_f16_vec8(const float* __restrict__ src,
                                _Float16* __restrict__ dst, int n8) {
  int i = blockIdx.x * blockDim.x + threadIdx.x;
  if (i >= n8) return;
  v4f a = *(const v4f*)(src + i * 8);
  v4f b = *(const v4f*)(src + i * 8 + 4);
  v8h o;
#pragma unroll
  for (int e = 0; e < 4; ++e) { o[e] = (_Float16)a[e]; o[e + 4] = (_Float16)b[e]; }
  *(v8h*)(dst + i * 8) = o;
}

// Wt[n*512 + k] = (f16) W[k*512 + n]  -- tiled transpose through LDS.
__global__ void w_transpose_f16(const float* __restrict__ W,
                                _Float16* __restrict__ Wt) {
  __shared__ float tile[16][17];
  const int tx = threadIdx.x, ty = threadIdx.y;
  const int k0 = blockIdx.y * 16, n0 = blockIdx.x * 16;
  tile[ty][tx] = W[(size_t)(k0 + ty) * DMODEL + n0 + tx];   // coalesced read
  __syncthreads();
  Wt[(size_t)(n0 + ty) * DMODEL + k0 + tx] = (_Float16)tile[tx][ty];
}

// C[M, n0..n0+63] strip per wave. A[M,K] f16 row-major, Bt[N,K] f16 (B transposed).
// mode 0: store f16 row-major (scaled)   -> Ch[m*N + n]
// mode 1: store f16 transposed (scaled)  -> Ch[n*M + m]   (for V^T)
// mode 2: store f32 row-major + bias     -> Cf[m*N + n]
__global__ void gemm_wmma(const _Float16* __restrict__ A, const _Float16* __restrict__ Bt,
                          _Float16* __restrict__ Ch, float* __restrict__ Cf,
                          const float* __restrict__ bias,
                          int M, int N, int K, float scale, int mode) {
  const int lane = threadIdx.x & 31;
  const int wave = (blockIdx.x * blockDim.x + threadIdx.x) >> 5;
  const int stripsN = N >> 6;                     // 64-wide N strips
  const int m0 = (wave / stripsN) << 4;
  const int n0 = (wave % stripsN) << 6;
  if (m0 >= M) return;
  const int lm = lane & 15;
  const int hi = lane >> 4;

  v8f acc[4] = {{}, {}, {}, {}};
  const _Float16* arow = A + (size_t)(m0 + lm) * K;
  const int kSteps = K >> 5;
  for (int kb = 0; kb < kSteps; ++kb) {
    const int d0 = kb * 32 + hi * 16;
    __builtin_prefetch(arow + d0 + 32, 0, 1);     // next k-step A chunk
    const v16h a = pack16(*(const v8h*)(arow + d0), *(const v8h*)(arow + d0 + 8));
    const int r0 = kb * 32 + hi * 16;
#pragma unroll
    for (int t = 0; t < 4; ++t) {
      // B fragment: lane owns col n0+t*16+lm, 16 consecutive K values -> one 32B load
      const v16h b = *(const v16h*)(Bt + (size_t)(n0 + t * 16 + lm) * K + r0);
      acc[t] = __builtin_amdgcn_wmma_f32_16x16x32_f16(false, a, false, b, (short)0,
                                                      acc[t], false, false);
    }
  }

  if (mode == 2) {
#pragma unroll
    for (int t = 0; t < 4; ++t) {
      const int n = n0 + t * 16 + lm;
      const float bv = bias ? bias[n] : 0.0f;
#pragma unroll
      for (int r = 0; r < 8; ++r) {
        const int m = m0 + r + hi * 8;
        Cf[(size_t)m * N + n] = acc[t][r] * scale + bv;
      }
    }
  } else if (mode == 1) {
#pragma unroll
    for (int t = 0; t < 4; ++t) {
      v8h o;
#pragma unroll
      for (int r = 0; r < 8; ++r) o[r] = (_Float16)(acc[t][r] * scale);
      *(v8h*)(Ch + (size_t)(n0 + t * 16 + lm) * M + m0 + hi * 8) = o;
    }
  } else {
#pragma unroll
    for (int t = 0; t < 4; ++t) {
#pragma unroll
      for (int r = 0; r < 8; ++r) {
        const int m = m0 + r + hi * 8;
        Ch[(size_t)m * N + n0 + t * 16 + lm] = (_Float16)(acc[t][r] * scale);
      }
    }
  }
}

// Flash attention, 64-key blocks: one wave per (head, 16-query-row block).
// Qh,Kh: [S, D] f16 row-major. Vt: [D, S] f16 (transposed). Ctx: [S, D] f16.
// Softmax scale is folded into Qh.
#define ATT_WAVES 4
__global__ void flash_attn_wmma(const _Float16* __restrict__ Qh,
                                const _Float16* __restrict__ Kh,
                                const _Float16* __restrict__ Vt,
                                _Float16* __restrict__ Ctx) {
  __shared__ __align__(16) _Float16 lds[ATT_WAVES][16][72];  // 64 keys + pad
  const int lane = threadIdx.x & 31;
  const int wv = threadIdx.x >> 5;
  const int wave = blockIdx.x * ATT_WAVES + wv;
  const int nq = S_LEN / 16;
  const int h = wave / nq;
  const int qb = wave % nq;
  if (h >= NHEADS) return;
  const int lm = lane & 15;
  const int hi = lane >> 4;
  const int m0 = qb << 4;

  // Q fragments for d=0..31 and d=32..63 (A layout)
  const _Float16* qrow = Qh + (size_t)(m0 + lm) * DMODEL + h * HDIM;
  const v16h qa0 = pack16(*(const v8h*)(qrow + hi * 16),
                          *(const v8h*)(qrow + hi * 16 + 8));
  const v16h qa1 = pack16(*(const v8h*)(qrow + 32 + hi * 16),
                          *(const v8h*)(qrow + 32 + hi * 16 + 8));

  v8f o[4] = {{}, {}, {}, {}};
  float mrow[8], lrow[8];
#pragma unroll
  for (int r = 0; r < 8; ++r) { mrow[r] = -1e30f; lrow[r] = 0.0f; }

  for (int j = 0; j < S_LEN / 64; ++j) {
    const int k0 = j * 64;

    // Scores for 4 groups of 16 keys: 8 WMMAs.
    v8f s[4] = {{}, {}, {}, {}};
#pragma unroll
    for (int t = 0; t < 4; ++t) {
      const _Float16* kr = Kh + (size_t)(k0 + t * 16 + lm) * DMODEL + h * HDIM + hi * 16;
      if (t == 0) __builtin_prefetch(kr + 64 * DMODEL, 0, 1);  // next key block
      s[t] = __builtin_amdgcn_wmma_f32_16x16x32_f16(false, qa0, false,
               *(const v16h*)(kr), (short)0, s[t], false, false);
      s[t] = __builtin_amdgcn_wmma_f32_16x16x32_f16(false, qa1, false,
               *(const v16h*)(kr + 32), (short)0, s[t], false, false);
    }

    // Online softmax over 64 keys: row r lives across the 16 lanes of this half-wave.
#pragma unroll
    for (int r = 0; r < 8; ++r) {
      float tm = fmaxf(fmaxf(s[0][r], s[1][r]), fmaxf(s[2][r], s[3][r]));
#pragma unroll
      for (int off = 1; off < 16; off <<= 1) tm = fmaxf(tm, __shfl_xor(tm, off, 32));
      const float mnew = fmaxf(mrow[r], tm);
      const float corr = __expf(mrow[r] - mnew);
      float p[4];
#pragma unroll
      for (int t = 0; t < 4; ++t) p[t] = __expf(s[t][r] - mnew);
      float rs = (p[0] + p[1]) + (p[2] + p[3]);
#pragma unroll
      for (int off = 1; off < 16; off <<= 1) rs += __shfl_xor(rs, off, 32);
      lrow[r] = lrow[r] * corr + rs;
      mrow[r] = mnew;
#pragma unroll
      for (int g = 0; g < 4; ++g) o[g][r] *= corr;
#pragma unroll
      for (int t = 0; t < 4; ++t)
        lds[wv][r + hi * 8][lm + t * 16] = (_Float16)p[t];   // C layout -> LDS
    }
    asm volatile("s_wait_dscnt 0x0" ::: "memory");   // intra-wave LDS RAW

    // Re-read P as two A-layout fragments (keys 0..31 and 32..63 of this block).
    const _Float16* prow = &lds[wv][lm][0];
    const v16h pa0 = pack16(*(const v8h*)(prow + hi * 16),
                            *(const v8h*)(prow + hi * 16 + 8));
    const v16h pa1 = pack16(*(const v8h*)(prow + 32 + hi * 16),
                            *(const v8h*)(prow + 32 + hi * 16 + 8));

    // P(16x64) x V(64x64): 8 WMMAs; Vt makes each B fragment one 32B load.
#pragma unroll
    for (int g = 0; g < 4; ++g) {
      const _Float16* vr = Vt + (size_t)(h * HDIM + g * 16 + lm) * S_LEN + k0 + hi * 16;
      o[g] = __builtin_amdgcn_wmma_f32_16x16x32_f16(false, pa0, false,
               *(const v16h*)(vr), (short)0, o[g], false, false);
      o[g] = __builtin_amdgcn_wmma_f32_16x16x32_f16(false, pa1, false,
               *(const v16h*)(vr + 32), (short)0, o[g], false, false);
    }
  }

#pragma unroll
  for (int r = 0; r < 8; ++r) {
    const float inv = 1.0f / lrow[r];
    const int m = m0 + r + hi * 8;
    _Float16* crow = Ctx + (size_t)m * DMODEL + h * HDIM;
#pragma unroll
    for (int g = 0; g < 4; ++g)
      crow[g * 16 + lm] = (_Float16)(o[g][r] * inv);
  }
}

extern "C" void kernel_launch(void* const* d_in, const int* in_sizes, int n_in,
                              void* d_out, int out_size, void* d_ws, size_t ws_size,
                              hipStream_t stream) {
  (void)in_sizes; (void)n_in; (void)out_size; (void)ws_size;
  const float* X    = (const float*)d_in[0];   // [1,4096,512]
  const float* Wq   = (const float*)d_in[1];   // [512,512]
  const float* Wk   = (const float*)d_in[2];
  const float* Wv   = (const float*)d_in[3];
  const float* Wo   = (const float*)d_in[4];
  const float* bout = (const float*)d_in[5];   // [512]
  float* out = (float*)d_out;

  const int NX = S_LEN * DMODEL;      // 2,097,152
  const int NW = DMODEL * DMODEL;     //   262,144

  _Float16* ws  = (_Float16*)d_ws;    // ~22 MB of f16 scratch
  _Float16* Xh  = ws;
  _Float16* Wqt = Xh  + NX;           // weights stored transposed [N][K]
  _Float16* Wkt = Wqt + NW;
  _Float16* Wvt = Wkt + NW;
  _Float16* Wot = Wvt + NW;
  _Float16* Qh  = Wot + NW;
  _Float16* Kh  = Qh  + NX;
  _Float16* Vt  = Kh  + NX;           // transposed [D, S]
  _Float16* Ctx = Vt  + NX;

  f32_to_f16_vec8<<<(NX / 8 + 255) / 256, 256, 0, stream>>>(X, Xh, NX / 8);

  dim3 tgrid(DMODEL / 16, DMODEL / 16), tblk(16, 16);
  w_transpose_f16<<<tgrid, tblk, 0, stream>>>(Wq, Wqt);
  w_transpose_f16<<<tgrid, tblk, 0, stream>>>(Wk, Wkt);
  w_transpose_f16<<<tgrid, tblk, 0, stream>>>(Wv, Wvt);
  w_transpose_f16<<<tgrid, tblk, 0, stream>>>(Wo, Wot);

  const int strips = (S_LEN / 16) * (DMODEL / 64); // 2048 waves (16x64 per wave)
  const int gemmBlocks = strips / 4;               // 4 waves (128 thr) per block
  const float qscale = 0.125f;                     // 1/sqrt(64) folded into Q
  gemm_wmma<<<gemmBlocks, 128, 0, stream>>>(Xh, Wqt, Qh, nullptr, nullptr,
                                            S_LEN, DMODEL, DMODEL, qscale, 0);
  gemm_wmma<<<gemmBlocks, 128, 0, stream>>>(Xh, Wkt, Kh, nullptr, nullptr,
                                            S_LEN, DMODEL, DMODEL, 1.0f, 0);
  gemm_wmma<<<gemmBlocks, 128, 0, stream>>>(Xh, Wvt, Vt, nullptr, nullptr,
                                            S_LEN, DMODEL, DMODEL, 1.0f, 1);

  const int attnWaves = NHEADS * (S_LEN / 16);     // 2048
  flash_attn_wmma<<<attnWaves / ATT_WAVES, ATT_WAVES * 32, 0, stream>>>(Qh, Kh, Vt, Ctx);

  gemm_wmma<<<gemmBlocks, 128, 0, stream>>>(Ctx, Wot, nullptr, out, bout,
                                            S_LEN, DMODEL, DMODEL, 1.0f, 2);
}
